// ConformerMhsaBlock_17102559772809
// MI455X (gfx1250) — matrix operand order repaired and checked by reading, not compiled, not run on hardware
//
#include <hip/hip_runtime.h>
#include <hip/hip_bf16.h>
#include <math.h>
#include <stdint.h>

// ---------------------------------------------------------------------------
// Conformer MHSA block for gfx1250 (MI455X), wave32 + WMMA bf16 path.
// B=16, T=1024, D=512, H=8, hd=64.  All matmuls on v_wmma_f32_16x16x32_bf16.
// Shared tiles (weights, K/V chunks) double-buffer-staged in LDS via
// GLOBAL_LOAD_ASYNC_TO_LDS_B128 with overlapped s_wait_asynccnt pipelining.
// ---------------------------------------------------------------------------

typedef __attribute__((ext_vector_type(16))) __bf16 v16bf;
typedef __attribute__((ext_vector_type(8)))  __bf16 v8bf;
typedef __attribute__((ext_vector_type(8)))  float  v8f;

#define BDIM 16
#define TDIM 1024
#define DDIM 512
#define HN   8
#define HDIM 64
#define MTOT (BDIM * TDIM)   // 16384 tokens

__device__ __forceinline__ unsigned short f2bf(float f) {
  union { float f; unsigned int u; } v; v.f = f;
  unsigned int r = v.u + 0x7FFFu + ((v.u >> 16) & 1u);   // round-to-nearest-even
  return (unsigned short)(r >> 16);
}

__device__ __forceinline__ v16bf load_frag16(const unsigned short* lo,
                                             const unsigned short* hi) {
  v8bf a = *(const v8bf*)lo;
  v8bf b = *(const v8bf*)hi;
  return __builtin_shufflevector(a, b, 0,1,2,3,4,5,6,7,8,9,10,11,12,13,14,15);
}

__device__ __forceinline__ v16bf load_fragB(const unsigned short* p, int ld, int half) {
  v16bf r;
#pragma unroll
  for (int s = 0; s < 8; ++s) {
    r[s]     = __builtin_bit_cast(__bf16, p[(8 * half + s) * ld]);
    r[8 + s] = __builtin_bit_cast(__bf16, p[(16 + 8 * half + s) * ld]);
  }
  return r;
}

__device__ __forceinline__ v8f zero8() {
  v8f z = {0.f,0.f,0.f,0.f,0.f,0.f,0.f,0.f};
  return z;
}

// Async 16-byte global -> LDS copy (GVS mode: SGPR64 base + per-lane byte off).
// Tracked on ASYNCcnt.  No "memory" clobber on the issue: ordering vs LDS
// consumers is enforced by the (volatile, clobbering) wait + __syncthreads().
__device__ __forceinline__ void async_ld_b128(const void* lds_dst,
                                              const void* gbase,
                                              unsigned byte_off) {
  const unsigned lds_off = (unsigned)(uintptr_t)lds_dst;  // low 32b = LDS addr
  asm volatile("global_load_async_to_lds_b128 %0, %1, %2"
               :
               : "v"(lds_off), "v"(byte_off),
                 "s"((unsigned long long)(uintptr_t)gbase));
}
// Wait until at most N async ops outstanding (async loads complete in order,
// so N = size of the in-flight next stage guarantees the current tile landed).
template <int N>
__device__ __forceinline__ void wait_async() {
  asm volatile("s_wait_asynccnt %0" :: "n"(N) : "memory");
}

// ---------------------------------------------------------------------------
// Kernel 1: convert the four 512x512 weight matrices f32 -> bf16 (no transpose:
// row-major [K_in, N] is exactly the WMMA B-fragment feed layout).
// ---------------------------------------------------------------------------
__global__ void __launch_bounds__(256)
convert_weights(const float* __restrict__ wq, const float* __restrict__ wk,
                const float* __restrict__ wv, const float* __restrict__ wo,
                unsigned short* __restrict__ oq, unsigned short* __restrict__ ok,
                unsigned short* __restrict__ ov, unsigned short* __restrict__ oo) {
  int i = blockIdx.x * 256 + threadIdx.x;
  if (i < DDIM * DDIM) {
    oq[i] = f2bf(wq[i]);
    ok[i] = f2bf(wk[i]);
    ov[i] = f2bf(wv[i]);
    oo[i] = f2bf(wo[i]);
  }
}

// ---------------------------------------------------------------------------
// Kernel 2: LayerNorm, one wave per 512-wide row, output bf16.
// ---------------------------------------------------------------------------
__global__ void __launch_bounds__(256)
ln_kernel(const float* __restrict__ x, const float* __restrict__ scale,
          const float* __restrict__ bias, unsigned short* __restrict__ y) {
  const int row  = blockIdx.x * 8 + (threadIdx.x >> 5);
  const int lane = threadIdx.x & 31;
  const float* xr = x + (size_t)row * DDIM;

  float v[16];
  float s = 0.f, ss = 0.f;
#pragma unroll
  for (int i = 0; i < 16; ++i) {
    v[i] = xr[lane + 32 * i];
    s  += v[i];
    ss += v[i] * v[i];
  }
#pragma unroll
  for (int m = 16; m >= 1; m >>= 1) {
    s  += __shfl_xor(s,  m, 32);
    ss += __shfl_xor(ss, m, 32);
  }
  const float mean = s * (1.f / DDIM);
  const float var  = ss * (1.f / DDIM) - mean * mean;
  const float inv  = rsqrtf(var + 1e-6f);

  unsigned short* yr = y + (size_t)row * DDIM;
#pragma unroll
  for (int i = 0; i < 16; ++i) {
    const int c = lane + 32 * i;
    yr[c] = f2bf((v[i] - mean) * inv * scale[c] + bias[c]);
  }
}

// ---------------------------------------------------------------------------
// Kernel 3/4/5/7: bf16 WMMA GEMM  C[M,512] = A[M,512] * W[512,512]  (+epilogue)
//   MODE 0: Q  = (C + bq) * (r_softplus0 * softplus(qs[n&63]) / 8)  -> bf16
//   MODE 1: K  = (C + bk)                -> bf16, stored transposed Kt[b,h,hd,t]
//   MODE 2: V  = (C + bv)                -> bf16, natural layout
//   MODE 3: out= C + bo + residual(x)    -> f32 (d_out)
// Block: 256 threads (8 waves); tile 128(M) x 128(N); wave owns 16x128.
// W k-tile (32x128 bf16 = 8KB) double-buffer async-staged in LDS; copy of
// k-step i+1 overlaps WMMA of step i.
// ---------------------------------------------------------------------------
template <int MODE>
__global__ void __launch_bounds__(256)
gemm512(const unsigned short* __restrict__ A,
        const unsigned short* __restrict__ W,
        const float* __restrict__ bias,
        const float* __restrict__ qsp,
        const float* __restrict__ resid,
        unsigned short* __restrict__ outb,
        float* __restrict__ outf) {
  __shared__ __align__(16) unsigned short WL[2][32 * 128];

  const int t    = threadIdx.x;
  const int lane = t & 31;
  const int wave = t >> 5;
  const int half = lane >> 4;
  const int l15  = lane & 15;
  const int m0   = blockIdx.x * 128 + wave * 16;
  const int n0   = blockIdx.y * 128;

  // stage one 32x128 tile: 2 x b128 per thread
  auto stage = [&](int buf, int k0) {
#pragma unroll
    for (int i = 0; i < 2; ++i) {
      const int e0  = (t + 256 * i) * 8;
      const int row = e0 >> 7;          // 0..31
      const int col = e0 & 127;         // multiples of 8
      async_ld_b128(&WL[buf][e0], W,
                    (unsigned)(((k0 + row) * DDIM + n0 + col) * 2));
    }
  };

  v8f acc[8];
#pragma unroll
  for (int nf = 0; nf < 8; ++nf) acc[nf] = zero8();

  const unsigned short* Arow = A + (size_t)(m0 + l15) * DDIM;

  stage(0, 0);
  for (int k0 = 0; k0 < DDIM; k0 += 32) {
    const int cur = (k0 >> 5) & 1;
    if (k0 + 32 < DDIM) {
      stage(cur ^ 1, k0 + 32);   // prefetch next tile into the other buffer
      wait_async<2>();           // current tile (2 older ops) has landed
    } else {
      wait_async<0>();
    }
    __syncthreads();

    const v16bf a = load_frag16(Arow + k0 + 8 * half,
                                Arow + k0 + 16 + 8 * half);
    const unsigned short* Brow = WL[cur] + l15;
#pragma unroll
    for (int nf = 0; nf < 8; ++nf) {
      const v16bf b = load_fragB(Brow + nf * 16, 128, half);
      acc[nf] = __builtin_amdgcn_wmma_f32_16x16x32_bf16(
          false, a, false, b, (short)0, acc[nf], false, false);
    }
    __syncthreads();   // all reads of WL[cur] done before it is restaged
  }

#pragma unroll
  for (int nf = 0; nf < 8; ++nf) {
    const int n = n0 + nf * 16 + l15;
    const float bs = bias[n];
    float scl = 1.0f;
    if (MODE == 0) {
      const float qp = qsp[n & (HDIM - 1)];
      scl = 1.442695041f * log1pf(__expf(qp)) * 0.125f;   // softplus / sqrt(64)
    }
#pragma unroll
    for (int r = 0; r < 8; ++r) {
      const int m = m0 + r + 8 * half;
      float val = acc[nf][r] + bs;
      if (MODE == 0) val *= scl;
      if (MODE == 3) {
        outf[(size_t)m * DDIM + n] = val + resid[(size_t)m * DDIM + n];
      } else if (MODE == 1) {
        const int bb = m >> 10, tt = m & (TDIM - 1);
        const int h = n >> 6, k = n & (HDIM - 1);
        outb[((size_t)(bb * HN + h) * HDIM + k) * TDIM + tt] = f2bf(val);
      } else {
        outb[(size_t)m * DDIM + n] = f2bf(val);
      }
    }
  }
}

// ---------------------------------------------------------------------------
// Kernel 6: flash-style attention.  Block = (b, h, 128 q-rows), 8 waves; each
// wave owns 16 q-rows, streams Tk in 64-key chunks.  Kt/V chunks (8KB each)
// double-buffer async-staged in LDS (copy of chunk i+1 overlaps chunk i).
// Q A-fragments preloaded into VGPRs.  S = Q*Kt via WMMA, f32 online softmax
// with half-wave shfl row-reductions, probs -> per-wave LDS bf16 tile
// (re-enters A-fragment layout via ds_load_b128), ctx += P*V via WMMA.
// ---------------------------------------------------------------------------
__global__ void __launch_bounds__(256)
attn_kernel(const unsigned short* __restrict__ Qb,   // [M,512] bf16 (scaled)
            const unsigned short* __restrict__ Ktb,  // [B*H*64, 1024] bf16
            const unsigned short* __restrict__ Vb,   // [M,512] bf16
            const float* __restrict__ pads,          // [B*T] (1.0 = padded)
            unsigned short* __restrict__ ctx) {      // [M,512] bf16
  __shared__ __align__(16) unsigned short Plds[8 * 16 * 64];  // 16KB probs
  __shared__ __align__(16) unsigned short KtL[2][64 * 64];    // [hd][key] 2x8KB
  __shared__ __align__(16) unsigned short VL[2][64 * 64];     // [key][hd] 2x8KB

  const int blk  = blockIdx.x;          // b*64 + h*8 + qblk
  const int qblk = blk & 7;
  const int h    = (blk >> 3) & 7;
  const int b    = blk >> 6;
  const int t    = threadIdx.x;
  const int lane = t & 31;
  const int wave = t >> 5;
  const int half = lane >> 4;
  const int l15  = lane & 15;
  const int q0   = qblk * 128 + wave * 16;

  unsigned short* Pw = Plds + wave * 16 * 64;

  const unsigned short* KtBase = Ktb + (size_t)(b * HN + h) * HDIM * TDIM;
  const unsigned short* Vbase  = Vb + (size_t)(b * TDIM) * DDIM + h * HDIM;

  // stage one 64-key chunk (Kt tile + V tile): 4 x b128 per thread
  auto stage = [&](int buf, int tk0) {
#pragma unroll
    for (int i = 0; i < 2; ++i) {
      const int e0  = (t + i * 256) * 8;   // element offset in 64x64 tile
      const int row = e0 >> 6;
      const int col = e0 & 63;
      async_ld_b128(&KtL[buf][e0], KtBase,
                    (unsigned)((row * TDIM + tk0 + col) * 2));
      async_ld_b128(&VL[buf][e0], Vbase,
                    (unsigned)(((tk0 + row) * DDIM + col) * 2));
    }
  };

  float qvalid[8];
#pragma unroll
  for (int r = 0; r < 8; ++r)
    qvalid[r] = (pads[b * TDIM + q0 + r + 8 * half] < 0.5f) ? 1.f : 0.f;

  // preload the two Q A-fragments (loop-invariant across key chunks)
  const unsigned short* Qrow =
      Qb + (size_t)(b * TDIM + q0 + l15) * DDIM + h * HDIM;
  v16bf aq[2];
#pragma unroll
  for (int kk = 0; kk < 2; ++kk)
    aq[kk] = load_frag16(Qrow + kk * 32 + 8 * half,
                         Qrow + kk * 32 + 16 + 8 * half);

  v8f acc[4];
#pragma unroll
  for (int nf = 0; nf < 4; ++nf) acc[nf] = zero8();
  float rmax[8], rsum[8];
#pragma unroll
  for (int r = 0; r < 8; ++r) { rmax[r] = -1e30f; rsum[r] = 0.f; }

  stage(0, 0);
  for (int tk0 = 0; tk0 < TDIM; tk0 += 64) {
    const int cur = (tk0 >> 6) & 1;
    if (tk0 + 64 < TDIM) {
      stage(cur ^ 1, tk0 + 64);   // prefetch next chunk into the other buffer
      wait_async<4>();            // current chunk (4 older ops) has landed
    } else {
      wait_async<0>();
    }
    __syncthreads();

    // ---- S tile: 16 q-rows x 64 keys ----
    v8f s[4];
#pragma unroll
    for (int nf = 0; nf < 4; ++nf) s[nf] = zero8();
#pragma unroll
    for (int kk = 0; kk < 2; ++kk) {
      const unsigned short* Br = KtL[cur] + kk * 32 * 64 + l15;
#pragma unroll
      for (int nf = 0; nf < 4; ++nf) {
        const v16bf bf = load_fragB(Br + nf * 16, 64, half);
        s[nf] = __builtin_amdgcn_wmma_f32_16x16x32_bf16(
            false, aq[kk], false, bf, (short)0, s[nf], false, false);
      }
    }

    // ---- mask + online softmax (f32) ----
    float kval[4];
#pragma unroll
    for (int nf = 0; nf < 4; ++nf)
      kval[nf] = (pads[b * TDIM + tk0 + nf * 16 + l15] < 0.5f) ? 1.f : 0.f;

    float nmax[8];
#pragma unroll
    for (int r = 0; r < 8; ++r) nmax[r] = rmax[r];
#pragma unroll
    for (int nf = 0; nf < 4; ++nf) {
#pragma unroll
      for (int r = 0; r < 8; ++r) {
        float sv = (kval[nf] > 0.f && qvalid[r] > 0.f) ? s[nf][r] : -1e30f;
        s[nf][r] = sv;
        nmax[r] = fmaxf(nmax[r], sv);
      }
    }
#pragma unroll
    for (int m = 8; m >= 1; m >>= 1) {   // reduce across the 16-lane half
#pragma unroll
      for (int r = 0; r < 8; ++r)
        nmax[r] = fmaxf(nmax[r], __shfl_xor(nmax[r], m, 32));
    }
#pragma unroll
    for (int r = 0; r < 8; ++r) {
      const float scl = __expf(rmax[r] - nmax[r]);
      rmax[r] = nmax[r];
      rsum[r] *= scl;
#pragma unroll
      for (int nf = 0; nf < 4; ++nf) acc[nf][r] *= scl;
    }
#pragma unroll
    for (int nf = 0; nf < 4; ++nf) {
#pragma unroll
      for (int r = 0; r < 8; ++r) {
        const float p = __expf(s[nf][r] - rmax[r]);
        rsum[r] += p;
        Pw[(r + 8 * half) * 64 + nf * 16 + l15] = f2bf(p);
      }
    }
    __syncthreads();   // probs visible; all KtL[cur] reads done

    // ---- ctx += P * V ----
#pragma unroll
    for (int tkk = 0; tkk < 64; tkk += 32) {
      const v16bf a = load_frag16(Pw + l15 * 64 + tkk + 8 * half,
                                  Pw + l15 * 64 + tkk + 16 + 8 * half);
      const unsigned short* Br = VL[cur] + tkk * 64 + l15;
#pragma unroll
      for (int nf = 0; nf < 4; ++nf) {
        const v16bf bf = load_fragB(Br + nf * 16, 64, half);
        acc[nf] = __builtin_amdgcn_wmma_f32_16x16x32_bf16(
            false, a, false, bf, (short)0, acc[nf], false, false);
      }
    }
    __syncthreads();   // VL[cur]/Plds reads done before cur is restaged
  }

  // ---- finalize: rows sum across half-wave, normalize, store ctx bf16 ----
#pragma unroll
  for (int m = 8; m >= 1; m >>= 1) {
#pragma unroll
    for (int r = 0; r < 8; ++r) rsum[r] += __shfl_xor(rsum[r], m, 32);
  }
#pragma unroll
  for (int r = 0; r < 8; ++r) rsum[r] = 1.0f / rsum[r];

#pragma unroll
  for (int nf = 0; nf < 4; ++nf) {
#pragma unroll
    for (int r = 0; r < 8; ++r) {
      const int m = b * TDIM + q0 + r + 8 * half;
      ctx[(size_t)m * DDIM + h * HDIM + nf * 16 + l15] =
          f2bf(acc[nf][r] * rsum[r]);
    }
  }
}

// ---------------------------------------------------------------------------
// Host-side launch
// ---------------------------------------------------------------------------
extern "C" void kernel_launch(void* const* d_in, const int* in_sizes, int n_in,
                              void* d_out, int out_size, void* d_ws, size_t ws_size,
                              hipStream_t stream) {
  const float* x    = (const float*)d_in[0];
  const float* pads = (const float*)d_in[1];
  const float* lns  = (const float*)d_in[2];
  const float* lnb  = (const float*)d_in[3];
  const float* wq   = (const float*)d_in[4];
  const float* bq   = (const float*)d_in[5];
  const float* wk   = (const float*)d_in[6];
  const float* bk   = (const float*)d_in[7];
  const float* wv   = (const float*)d_in[8];
  const float* bv   = (const float*)d_in[9];
  const float* wo   = (const float*)d_in[10];
  const float* bo   = (const float*)d_in[11];
  const float* qs   = (const float*)d_in[12];
  float* out = (float*)d_out;

  unsigned char* ws = (unsigned char*)d_ws;
  const size_t WSZ = (size_t)DDIM * DDIM * sizeof(unsigned short);  // 512 KB
  const size_t MSZ = (size_t)MTOT * DDIM * sizeof(unsigned short);  // 16 MB
  unsigned short* Wqb = (unsigned short*)(ws);              ws += WSZ;
  unsigned short* Wkb = (unsigned short*)(ws);              ws += WSZ;
  unsigned short* Wvb = (unsigned short*)(ws);              ws += WSZ;
  unsigned short* Wob = (unsigned short*)(ws);              ws += WSZ;
  unsigned short* Yb  = (unsigned short*)(ws);              ws += MSZ;
  unsigned short* Qbw = (unsigned short*)(ws);              ws += MSZ;
  unsigned short* Ktb = (unsigned short*)(ws);              ws += MSZ;
  unsigned short* Vbw = (unsigned short*)(ws);              ws += MSZ;
  unsigned short* Ctx = (unsigned short*)(ws);              ws += MSZ;

  convert_weights<<<(DDIM * DDIM + 255) / 256, 256, 0, stream>>>(
      wq, wk, wv, wo, Wqb, Wkb, Wvb, Wob);
  ln_kernel<<<MTOT / 8, 256, 0, stream>>>(x, lns, lnb, Yb);

  dim3 gg(MTOT / 128, DDIM / 128);
  gemm512<0><<<gg, 256, 0, stream>>>(Yb, Wqb, bq, qs, nullptr, Qbw, nullptr);
  gemm512<1><<<gg, 256, 0, stream>>>(Yb, Wkb, bk, nullptr, nullptr, Ktb, nullptr);
  gemm512<2><<<gg, 256, 0, stream>>>(Yb, Wvb, bv, nullptr, nullptr, Vbw, nullptr);

  attn_kernel<<<BDIM * HN * (TDIM / 128), 256, 0, stream>>>(
      Qbw, Ktb, Vbw, pads, Ctx);

  gemm512<3><<<gg, 256, 0, stream>>>(Ctx, Wob, bo, nullptr, x, nullptr, out);
}
